// PhenoCapsuleHead_48412871360756
// MI455X (gfx1250) — matrix operand order repaired and checked by reading, not compiled, not run on hardware
//
#include <hip/hip_runtime.h>
#include <cstdint>
#include <cstddef>

// ---------------------------------------------------------------------------
// Types for CDNA5 WMMA (wave32)
// ---------------------------------------------------------------------------
typedef __bf16 bf16;
typedef __attribute__((ext_vector_type(16))) __bf16 v16bf;
typedef __attribute__((ext_vector_type(8)))  __bf16 v8bf;
typedef __attribute__((ext_vector_type(8)))  float  v8f;

// Problem constants
#define BATCH 4096
#define DIN   2048
#define RR    7
#define DP    128
#define KK    25
#define DC    16
#define RDP   (RR*DP)        // 896
#define KH    (KK*DC)        // 400
#define RDC   (RR*DC)        // 112
#define NH1   (4*RR*DC)      // 448

// Output layout (flat, reference tuple order)
static constexpr size_t O_LOGITS = 0;                       // [B,K]        102400
static constexpr size_t O_CEMB   = 102400;                  // [B,K,112]    11468800
static constexpr size_t O_CPOSE  = 11571200;                // [B,K,16]     1638400
static constexpr size_t O_CACT   = 13209600;                // [B,K]        102400
static constexpr size_t O_Q      = 13312000;                // [B,R,K]      716800
static constexpr size_t O_POSE   = 14028800;                // [B,R,128]    3670016
static constexpr size_t O_ACT    = 17698816;                // [B,R]        28672
static constexpr size_t O_RCE    = 17727488;                // [B,R,K,16]   11468800

// Workspace layout (bytes)
static constexpr size_t W_ZB    = 0;                        // bf16 z        [4096][2048]
static constexpr size_t W_WPB   = 16777216;                 // bf16 W_primT  [896][2048]
static constexpr size_t W_POSEB = 20447232;                 // bf16 pose     [4096][896]
static constexpr size_t W_WCB   = 27787264;                 // bf16 w_capsT  [7][400][128]
static constexpr size_t W_WH1B  = 28504064;                 // bf16 W_h1T    [448][128] (K pad)
static constexpr size_t W_XB    = 28618752;                 // bf16 LN(cemb) [102400][128] (K pad)
// total: 54,833,152 bytes

// ---------------------------------------------------------------------------
// Helpers
// ---------------------------------------------------------------------------
static __device__ __forceinline__ bf16 f2bf(float f) {
    unsigned u = __float_as_uint(f);
    u += 0x7FFFu + ((u >> 16) & 1u);                 // round-to-nearest-even
    unsigned short h = (unsigned short)(u >> 16);
    return __builtin_bit_cast(bf16, h);
}

static __device__ __forceinline__ v8f wmma_bf16(v16bf a, v16bf b, v8f c) {
    return __builtin_amdgcn_wmma_f32_16x16x32_bf16(
        /*neg_a=*/false, a, /*neg_b=*/false, b,
        /*c_mod=*/(short)0, c, /*reuse_a=*/false, /*reuse_b=*/false);
}

// A fragment: 16x32 bf16 tile, row-major source (row = batch index), ld elems.
// lane<16 : row=lane,   K = {k0..k0+7, k0+16..k0+23}
// lane>=16: row=lane-16, K = {k0+8..k0+15, k0+24..k0+31}
static __device__ __forceinline__ v16bf load_a_frag(const bf16* __restrict__ A,
                                                    int ld, int k0) {
    int lane = threadIdx.x & 31;
    int row  = lane & 15;
    int kh   = (lane & 16) ? 8 : 0;
    const bf16* p = A + (size_t)row * ld + k0 + kh;
    v8bf lo = *(const v8bf*)p;
    v8bf hi = *(const v8bf*)(p + 16);
    v16bf r;
#pragma unroll
    for (int i = 0; i < 8; i++) { r[i] = lo[i]; r[8 + i] = hi[i]; }
    return r;
}

// B fragment: 32x16 bf16 (K x N) from N-major storage Bt[n][k], ld elems.
// lane<16 : col=lane,   K = {k0..k0+7, k0+16..k0+23} ; lane>=16: col=lane-16, K = {k0+8..k0+15, k0+24..k0+31}
static __device__ __forceinline__ v16bf load_b_frag(const bf16* __restrict__ Bt,
                                                    int ld, int k0) {
    int lane = threadIdx.x & 31;
    int n    = lane & 15;
    int kb   = (lane & 16) ? 8 : 0;
    const bf16* p = Bt + (size_t)n * ld + k0 + kb;
    v8bf lo = *(const v8bf*)p;
    v8bf hi = *(const v8bf*)(p + 16);
    v16bf r;
#pragma unroll
    for (int i = 0; i < 8; i++) { r[i] = lo[i]; r[8 + i] = hi[i]; }
    return r;
}

// ---------------------------------------------------------------------------
// Conversion / transpose kernels (one-time, memory-bound)
// ---------------------------------------------------------------------------
__global__ void __launch_bounds__(256) k_cvt_z(const float* __restrict__ z,
                                               bf16* __restrict__ zb) {
    size_t i = (size_t)blockIdx.x * 256 + threadIdx.x;
    if (i < (size_t)BATCH * DIN) zb[i] = f2bf(z[i]);
}

__global__ void __launch_bounds__(256) k_cvt_wprim(const float* __restrict__ W,
                                                   bf16* __restrict__ wpb) {
    size_t i = (size_t)blockIdx.x * 256 + threadIdx.x;
    if (i < (size_t)DIN * RDP) {
        int k = (int)(i / RDP), n = (int)(i % RDP);
        wpb[(size_t)n * DIN + k] = f2bf(W[i]);
    }
}

__global__ void __launch_bounds__(256) k_cvt_wcaps(const float* __restrict__ W,
                                                   bf16* __restrict__ wcb) {
    size_t i = (size_t)blockIdx.x * 256 + threadIdx.x;
    if (i < (size_t)RR * DP * KH) {
        int r   = (int)(i / (DP * KH));
        int rem = (int)(i % (DP * KH));
        int d   = rem / KH;
        int kh  = rem % KH;
        wcb[((size_t)r * KH + kh) * DP + d] = f2bf(W[i]);
    }
}

__global__ void __launch_bounds__(256) k_cvt_wh1(const float* __restrict__ W,
                                                 bf16* __restrict__ wh1b) {
    size_t i = (size_t)blockIdx.x * 256 + threadIdx.x;
    if (i < (size_t)NH1 * 128) {
        int n = (int)(i / 128), kk = (int)(i % 128);
        wh1b[i] = (kk < RDC) ? f2bf(W[(size_t)kk * NH1 + n]) : f2bf(0.f);
    }
}

// ---------------------------------------------------------------------------
// Kernel 1: pose = squash(z @ W_prim + b_prim)     [M=4096, K=2048, N=896]
// Tile: 128x128 per WG (8 waves x 16 rows); N-tile == one r-group so squash
// (row reduction over 128 cols) is fused. B tile double-buffered in LDS:
// next tile's global loads issued before the WMMA block (latency hidden),
// one barrier per k-step. All 8 B-fragments batched before the 8 WMMAs so
// the XDL pipe gets back-to-back issues.
// dynamic LDS: lB 2x12288B + lC 66048B + red 1024B + scale 512B = 92160B
// ---------------------------------------------------------------------------
__global__ void __launch_bounds__(256) k_gemm_pose(
        const bf16* __restrict__ zb, const bf16* __restrict__ wpb,
        const float* __restrict__ b_prim,
        float* __restrict__ pose_out, bf16* __restrict__ poseb) {
    extern __shared__ char smem1[];
    bf16*  lB   = (bf16*)smem1;                          // [2][128][48]
    float* lC   = (float*)(smem1 + 24576);               // [128][129]
    float* red  = (float*)(smem1 + 24576 + 66048);       // [256]
    float* scl  = red + 256;                             // [128]

    const int mBase = blockIdx.x * 128;
    const int r     = blockIdx.y;
    const int nBase = r * 128;
    const int tid   = threadIdx.x;
    const int wave  = tid >> 5;

    v8f acc[8];
#pragma unroll
    for (int i = 0; i < 8; i++)
#pragma unroll
        for (int j = 0; j < 8; j++) acc[i][j] = 0.f;

    const bf16* Arow = zb + (size_t)(mBase + wave * 16) * DIN;

    // cooperative B staging: thread -> (n = tid/2, kp = (tid&1)*16)
    const int n  = tid >> 1;
    const int kp = (tid & 1) * 16;
    const bf16* bsrc  = wpb + (size_t)(nBase + n) * DIN + kp;
    bf16* const bdst  = lB + n * 48 + kp;

    constexpr int NT = DIN / 32;   // 64 k-steps

    // preload tile 0 into buffer 0
    {
        v8bf x0 = *(const v8bf*)(bsrc);
        v8bf x1 = *(const v8bf*)(bsrc + 8);
        *(v8bf*)(bdst)     = x0;
        *(v8bf*)(bdst + 8) = x1;
    }
    __syncthreads();

    for (int kt = 0; kt < NT; kt++) {
        const int k0  = kt * 32;
        const int cur = kt & 1;
        const bool more = (kt + 1 < NT);

        // issue next B tile's global loads early (latency hidden by WMMAs)
        v8bf x0, x1;
        if (more) {
            const bf16* s = bsrc + (size_t)(kt + 1) * 32;
            x0 = *(const v8bf*)s;
            x1 = *(const v8bf*)(s + 8);
        }
        if (kt + 2 < NT)
            __builtin_prefetch(bsrc + (size_t)(kt + 2) * 32, 0, 3);

        // compute on current buffer: batch all loads, then 8 WMMAs
        const bf16* bufc = lB + cur * (128 * 48);
        v16bf a = load_a_frag(Arow, DIN, k0);
        v16bf bfr[8];
#pragma unroll
        for (int ns = 0; ns < 8; ns++)
            bfr[ns] = load_b_frag(bufc + ns * 16 * 48, 48, 0);
#pragma unroll
        for (int ns = 0; ns < 8; ns++)
            acc[ns] = wmma_bf16(a, bfr[ns], acc[ns]);

        if (more) {
            bf16* d = lB + (1 - cur) * (128 * 48) + n * 48 + kp;
            *(v8bf*)d       = x0;
            *(v8bf*)(d + 8) = x1;
        }
        __syncthreads();
    }

    // bias + spill fp32 tile to LDS
    const int lane = tid & 31;
#pragma unroll
    for (int ns = 0; ns < 8; ns++) {
        int col  = ns * 16 + (lane & 15);
        float bv = b_prim[nBase + col];
        int rb   = wave * 16 + 8 * (lane >> 4);
#pragma unroll
        for (int v = 0; v < 8; v++)
            lC[(rb + v) * 129 + col] = acc[ns][v] + bv;
    }
    __syncthreads();

    // squash: per-row (=per pose vector) sum of squares over 128 cols
    {
        int row = tid >> 1, c0 = (tid & 1) * 64;
        float s = 0.f;
        for (int j = 0; j < 64; j++) { float x = lC[row * 129 + c0 + j]; s += x * x; }
        red[tid] = s;
    }
    __syncthreads();
    if (tid < 128) {
        float ns_   = red[2 * tid] + red[2 * tid + 1];
        float scale = ns_ / fmaxf(1.f + ns_, 1e-6f);
        scl[tid]    = scale / (sqrtf(ns_) + 1e-6f);
    }
    __syncthreads();
    for (int idx = tid; idx < 128 * 128; idx += 256) {
        int row = idx >> 7, col = idx & 127;
        float v = lC[row * 129 + col] * scl[row];
        size_t o = ((size_t)(mBase + row) * RR + r) * DP + col;
        pose_out[o] = v;
        poseb[o]    = f2bf(v);
    }
}

// ---------------------------------------------------------------------------
// Kernel 2: act = clip(sigmoid(LN(z) @ W_act + b_act))   (memory-bound)
// One workgroup per batch row.
// ---------------------------------------------------------------------------
__global__ void __launch_bounds__(256) k_act(
        const float* __restrict__ z, const float* __restrict__ g,
        const float* __restrict__ bb, const float* __restrict__ W_act,
        const float* __restrict__ b_act, float* __restrict__ act_out) {
    __shared__ float zrow[DIN];
    __shared__ float red[256], red2[256];
    __shared__ float acc7[256 * 8];
    const int b = blockIdx.x, tid = threadIdx.x;
    const float* zp = z + (size_t)b * DIN;
    float s = 0.f, s2 = 0.f;
    for (int j = tid; j < DIN; j += 256) {
        float v = zp[j]; zrow[j] = v; s += v; s2 += v * v;
    }
    red[tid] = s; red2[tid] = s2;
    __syncthreads();
    for (int st = 128; st > 0; st >>= 1) {
        if (tid < st) { red[tid] += red[tid + st]; red2[tid] += red2[tid + st]; }
        __syncthreads();
    }
    float mean = red[0] * (1.f / DIN);
    float var  = red2[0] * (1.f / DIN) - mean * mean;
    float rstd = rsqrtf(var + 1e-5f);
    float la[7];
#pragma unroll
    for (int r = 0; r < 7; r++) la[r] = 0.f;
    for (int j = tid; j < DIN; j += 256) {
        float y = (zrow[j] - mean) * rstd * g[j] + bb[j];
        const float* w = W_act + (size_t)j * RR;
#pragma unroll
        for (int r = 0; r < 7; r++) la[r] += y * w[r];
    }
#pragma unroll
    for (int r = 0; r < 7; r++) acc7[tid * 8 + r] = la[r];
    __syncthreads();
    if (tid < 7) {
        float t = 0.f;
        for (int i = 0; i < 256; i++) t += acc7[i * 8 + tid];
        t += b_act[tid];
        float sg = 1.f / (1.f + expf(-t));
        act_out[(size_t)b * RR + tid] = fminf(fmaxf(sg, 0.f), 1.f);
    }
}

// ---------------------------------------------------------------------------
// Kernel 3: votes (WMMA) + 3 routing iterations fully fused, u lives in LDS.
// 16 batch rows per WG. dynamic LDS = (44800+2800+6400+128)*4 = 216512 bytes.
// ---------------------------------------------------------------------------
__global__ void __launch_bounds__(256) k_route(
        const bf16* __restrict__ poseb, const bf16* __restrict__ wcb,
        const float* __restrict__ act_out,
        const float* __restrict__ ln_c_g, const float* __restrict__ ln_c_b,
        const float* __restrict__ beta_u,
        const float* __restrict__ ln_h_g, const float* __restrict__ ln_h_b,
        float* __restrict__ out, bf16* __restrict__ xb) {
    extern __shared__ float sm[];
    float* uS  = sm;                 // [16][2800]  u(b,r*400+k*16+h)
    float* qS  = uS + 16 * 2800;     // [16][175]   q(b,r*25+k)
    float* cpS = qS + 16 * 175;      // [16][400]   class_pose(b,k*16+h)
    float* aS  = cpS + 16 * 400;     // [16][7]     act tile (pad to 128)

    const int b0   = blockIdx.x * 16;
    const int tid  = threadIdx.x;
    const int wave = tid >> 5;
    const int lane = tid & 31;

    if (tid < 112) aS[tid] = act_out[(size_t)b0 * RR + tid];
    for (int i = tid; i < 16 * 175; i += 256) qS[i] = 1.f / KK;

    // --- Phase A: votes via WMMA, 175 (r,ntile) tiles over 8 waves ---------
    for (int t = wave; t < RR * KK; t += 8) {
        int r = t / KK, nt = t % KK;
        v8f acc;
#pragma unroll
        for (int j = 0; j < 8; j++) acc[j] = 0.f;
        const bf16* A  = poseb + (size_t)b0 * RDP + r * DP;
        const bf16* Bt = wcb + ((size_t)r * KH + nt * DC) * DP;
        v16bf af[4], bfr[4];
#pragma unroll
        for (int kk = 0; kk < 4; kk++) af[kk]  = load_a_frag(A, RDP, kk * 32);
#pragma unroll
        for (int kk = 0; kk < 4; kk++) bfr[kk] = load_b_frag(Bt, DP, kk * 32);
#pragma unroll
        for (int kk = 0; kk < 4; kk++) acc = wmma_bf16(af[kk], bfr[kk], acc);
        int bl  = 8 * (lane >> 4);
        int col = nt * DC + (lane & 15);
#pragma unroll
        for (int v = 0; v < 8; v++)
            uS[(bl + v) * 2800 + r * KH + col] = acc[v];
    }
    __syncthreads();

    // --- Phase B: 3 routing iterations, all in LDS -------------------------
    const float scale = 0.25f;  // 1/sqrt(D_C)
    for (int it = 0; it < 3; it++) {
        // class_pose raw sums: (b,k,h) = sum_r q*act*u
        for (int i = tid; i < 16 * 400; i += 256) {
            int b = i / 400, rem = i % 400, k = rem >> 4;
            float s = 0.f;
#pragma unroll
            for (int r = 0; r < RR; r++)
                s += qS[b * 175 + r * KK + k] * aS[b * RR + r] *
                     uS[b * 2800 + r * KH + rem];
            cpS[i] = s;
        }
        __syncthreads();
        // LayerNorm over h (16) per (b,k)
        for (int i = tid; i < 16 * KK; i += 256) {
            float* p = cpS + i * DC;
            float m = 0.f;
#pragma unroll
            for (int h = 0; h < DC; h++) m += p[h];
            m *= (1.f / DC);
            float v = 0.f;
#pragma unroll
            for (int h = 0; h < DC; h++) { float d = p[h] - m; v += d * d; }
            v *= (1.f / DC);
            float rs = rsqrtf(v + 1e-5f);
#pragma unroll
            for (int h = 0; h < DC; h++)
                p[h] = (p[h] - m) * rs * ln_c_g[h] + ln_c_b[h];
        }
        __syncthreads();
        // agree (overwrites qS in place)
        for (int i = tid; i < 16 * 175; i += 256) {
            int b = i / 175, rem = i % 175, r = rem / KK, k = rem % KK;
            const float* cp = cpS + b * 400 + k * DC;
            const float* up = uS + b * 2800 + r * KH + k * DC;
            float s = 0.f;
#pragma unroll
            for (int h = 0; h < DC; h++) s += cp[h] * up[h];
            qS[i] = s * scale;
        }
        __syncthreads();
        // softmax over k per (b,r)
        if (tid < 112) {
            float* p = qS + tid * KK;
            float mx = -1e30f;
#pragma unroll
            for (int k = 0; k < KK; k++) mx = fmaxf(mx, p[k]);
            float ss = 0.f;
#pragma unroll
            for (int k = 0; k < KK; k++) { float e = expf(p[k] - mx); p[k] = e; ss += e; }
            float inv = 1.f / ss;
#pragma unroll
            for (int k = 0; k < KK; k++) p[k] *= inv;
        }
        __syncthreads();
    }

    // --- Outputs -----------------------------------------------------------
    for (int i = tid; i < 16 * 400; i += 256) {            // class_pose
        int b = i / 400, rem = i % 400;
        out[O_CPOSE + (size_t)(b0 + b) * 400 + rem] = cpS[i];
    }
    for (int i = tid; i < 16 * KK; i += 256) {             // class_act
        int b = i / KK, k = i % KK;
        float s = 0.f;
#pragma unroll
        for (int r = 0; r < RR; r++) s += qS[b * 175 + r * KK + k] * aS[b * RR + r];
        s += beta_u[k];
        out[O_CACT + (size_t)(b0 + b) * KK + k] = 1.f / (1.f + expf(-s));
    }
    for (int i = tid; i < 16 * 175; i += 256) {            // q
        int b = i / 175, rem = i % 175;
        out[O_Q + (size_t)(b0 + b) * 175 + rem] = qS[i];
    }
    for (int i = tid; i < 16 * 2800; i += 256) {           // rce + class_embed
        int b = i / 2800, rem = i % 2800;
        int r = rem / KH, kh = rem % KH, k = kh >> 4, h = kh & 15;
        float v = qS[b * 175 + r * KK + k] * uS[i];
        out[O_RCE + (size_t)(b0 + b) * 2800 + rem] = v;
        out[O_CEMB + ((size_t)(b0 + b) * KK + k) * RDC + r * DC + h] = v;
    }
    // xb = bf16(LN(class_embed)), K padded to 128, per (b,k) row
    for (int i = tid; i < 16 * KK; i += 256) {
        int b = i / KK, k = i % KK;
        float m = 0.f, s2 = 0.f;
        for (int j = 0; j < RDC; j++) {
            int r = j >> 4, h = j & 15;
            float v = qS[b * 175 + r * KK + k] * uS[b * 2800 + r * KH + k * DC + h];
            m += v; s2 += v * v;
        }
        m *= (1.f / RDC);
        float var = s2 * (1.f / RDC) - m * m;
        float rs  = rsqrtf(var + 1e-5f);
        bf16* dst = xb + ((size_t)(b0 + b) * KK + k) * 128;
        for (int j = 0; j < RDC; j++) {
            int r = j >> 4, h = j & 15;
            float v = qS[b * 175 + r * KK + k] * uS[b * 2800 + r * KH + k * DC + h];
            dst[j] = f2bf((v - m) * rs * ln_h_g[j] + ln_h_b[j]);
        }
        for (int j = RDC; j < 128; j++) dst[j] = f2bf(0.f);
    }
}

// ---------------------------------------------------------------------------
// Kernel 4: logits = gelu(LN(cemb) @ W_h1 + b_h1) @ W_h2 + b_h2
// M=102400 rows, K=128(pad), N=448; h never materialized.
// ---------------------------------------------------------------------------
__global__ void __launch_bounds__(256) k_mlp(
        const bf16* __restrict__ xb, const bf16* __restrict__ wh1b,
        const float* __restrict__ b_h1, const float* __restrict__ W_h2,
        const float* __restrict__ b_h2, float* __restrict__ logits) {
    const int tid = threadIdx.x, wave = tid >> 5, lane = tid & 31;
    const size_t m0 = (size_t)blockIdx.x * 128 + wave * 16;
    const bf16* A = xb + m0 * 128;

    v16bf afr[4];
#pragma unroll
    for (int kk = 0; kk < 4; kk++) afr[kk] = load_a_frag(A, 128, kk * 32);

    float part[8];
#pragma unroll
    for (int v = 0; v < 8; v++) part[v] = 0.f;

    for (int nt = 0; nt < NH1 / 16; nt++) {
        v8f acc;
#pragma unroll
        for (int j = 0; j < 8; j++) acc[j] = 0.f;
        const bf16* Bt = wh1b + (size_t)nt * 16 * 128;
        v16bf bfr[4];
#pragma unroll
        for (int kk = 0; kk < 4; kk++) bfr[kk] = load_b_frag(Bt, 128, kk * 32);
#pragma unroll
        for (int kk = 0; kk < 4; kk++) acc = wmma_bf16(afr[kk], bfr[kk], acc);
        int col   = nt * 16 + (lane & 15);
        float bia = b_h1[col];
        float w2  = W_h2[col];
#pragma unroll
        for (int v = 0; v < 8; v++) {
            float h = acc[v] + bia;
            float gl = 0.5f * h * (1.f + erff(h * 0.70710678118654752f));
            part[v] += gl * w2;
        }
    }
    float bh2 = b_h2[0];
#pragma unroll
    for (int v = 0; v < 8; v++) {
        float s = part[v];
        s += __shfl_xor(s, 1, 32);
        s += __shfl_xor(s, 2, 32);
        s += __shfl_xor(s, 4, 32);
        s += __shfl_xor(s, 8, 32);
        if ((lane & 15) == 0)
            logits[m0 + v + 8 * (lane >> 4)] = s + bh2;
    }
}

// ---------------------------------------------------------------------------
// Host launcher
// ---------------------------------------------------------------------------
extern "C" void kernel_launch(void* const* d_in, const int* in_sizes, int n_in,
                              void* d_out, int out_size, void* d_ws, size_t ws_size,
                              hipStream_t stream) {
    const float* z      = (const float*)d_in[0];
    const float* W_prim = (const float*)d_in[1];
    const float* b_prim = (const float*)d_in[2];
    const float* ln_a_g = (const float*)d_in[3];
    const float* ln_a_b = (const float*)d_in[4];
    const float* W_act  = (const float*)d_in[5];
    const float* b_act  = (const float*)d_in[6];
    const float* w_caps = (const float*)d_in[7];
    const float* ln_c_g = (const float*)d_in[8];
    const float* ln_c_b = (const float*)d_in[9];
    const float* beta_u = (const float*)d_in[10];
    const float* ln_h_g = (const float*)d_in[11];
    const float* ln_h_b = (const float*)d_in[12];
    const float* W_h1   = (const float*)d_in[13];
    const float* b_h1   = (const float*)d_in[14];
    const float* W_h2   = (const float*)d_in[15];
    const float* b_h2   = (const float*)d_in[16];

    float* out = (float*)d_out;
    char*  ws  = (char*)d_ws;
    bf16* zb    = (bf16*)(ws + W_ZB);
    bf16* wpb   = (bf16*)(ws + W_WPB);
    bf16* poseb = (bf16*)(ws + W_POSEB);
    bf16* wcb   = (bf16*)(ws + W_WCB);
    bf16* wh1b  = (bf16*)(ws + W_WH1B);
    bf16* xb    = (bf16*)(ws + W_XB);

    // conversions / transposes
    k_cvt_z    <<<(BATCH * DIN + 255) / 256, 256, 0, stream>>>(z, zb);
    k_cvt_wprim<<<(DIN * RDP + 255) / 256, 256, 0, stream>>>(W_prim, wpb);
    k_cvt_wcaps<<<(RR * DP * KH + 255) / 256, 256, 0, stream>>>(w_caps, wcb);
    k_cvt_wh1  <<<(NH1 * 128 + 255) / 256, 256, 0, stream>>>(W_h1, wh1b);

    // pose GEMM + squash (double-buffered B tile)
    size_t shm1 = 24576 + 66048 + 1024 + 512;
    (void)hipFuncSetAttribute((const void*)k_gemm_pose,
                              hipFuncAttributeMaxDynamicSharedMemorySize, (int)shm1);
    k_gemm_pose<<<dim3(BATCH / 128, RR), 256, shm1, stream>>>(
        zb, wpb, b_prim, out + O_POSE, poseb);

    // activations
    k_act<<<BATCH, 256, 0, stream>>>(z, ln_a_g, ln_a_b, W_act, b_act, out + O_ACT);

    // fused votes + routing (u resident in LDS: 216.5 KB, CDNA5 WGP has 320 KB)
    size_t shm3 = (size_t)(16 * 2800 + 16 * 175 + 16 * 400 + 128) * sizeof(float);
    (void)hipFuncSetAttribute((const void*)k_route,
                              hipFuncAttributeMaxDynamicSharedMemorySize, (int)shm3);
    k_route<<<BATCH / 16, 256, shm3, stream>>>(
        poseb, wcb, out + O_ACT, ln_c_g, ln_c_b, beta_u, ln_h_g, ln_h_b, out, xb);

    // MLP head -> logits
    k_mlp<<<(BATCH * KK) / 128, 256, 0, stream>>>(
        xb, wh1b, b_h1, W_h2, b_h2, out + O_LOGITS);
}
